// TopKRouter_69441031241774
// MI455X (gfx1250) — matrix operand order repaired and checked by reading, not compiled, not run on hardware
//
#include <hip/hip_runtime.h>
#include <hip/hip_bf16.h>
#include <math.h>
#include <stdint.h>

typedef __attribute__((ext_vector_type(16))) __bf16 v16bf;
typedef __attribute__((ext_vector_type(8)))  float  v8f;
typedef __attribute__((ext_vector_type(4)))  unsigned int v4u;
typedef __attribute__((ext_vector_type(8)))  int    v8i;
typedef __attribute__((ext_vector_type(4)))  int    v4i;

#define TOKENS   16384
#define D_IN     2048
#define D_HID    1024
#define N_EXP    64
#define TOP_K    8
#define SLOPE    0.01f

// ---------- helpers ----------
__device__ __forceinline__ unsigned short f2bf(float f) {
    unsigned u = __builtin_bit_cast(unsigned, f);
    unsigned r = u + 0x7fffu + ((u >> 16) & 1u);   // round-to-nearest-even
    return (unsigned short)(r >> 16);
}
__device__ __forceinline__ unsigned pack2bf(float lo, float hi) {
    return (unsigned)f2bf(lo) | ((unsigned)f2bf(hi) << 16);
}

union Frag {
    v16bf v;
    uint4 u[2];
};

// Tensor-DMA load (TDM): 2D tile of bf16 from global -> LDS (packed rows).
__device__ __forceinline__ void tdm_load_2d_bf16(unsigned ldsOff, const void* gptr,
                                                 unsigned tdim0, unsigned tdim1,
                                                 unsigned tile0, unsigned tile1,
                                                 unsigned ld) {
    unsigned long long ga = (unsigned long long)(uintptr_t)gptr;
    v4u g0;
    g0.x = 1u;                                        // count=1, user descriptor
    g0.y = ldsOff;                                    // lds_addr (bytes)
    g0.z = (unsigned)ga;                              // global_addr[31:0]
    g0.w = ((unsigned)(ga >> 32) & 0x1FFFFFFu) | 0x80000000u; // addr[56:32] | type=2
    v8i g1;
    g1[0] = (int)(1u << 16);                          // data_size=1 (2B); mask=0
    g1[1] = (int)((tdim0 & 0xFFFFu) << 16);           // tensor_dim0[15:0] @bit48
    g1[2] = (int)((tdim0 >> 16) | ((tdim1 & 0xFFFFu) << 16));
    g1[3] = (int)((tdim1 >> 16) | (tile0 << 16));
    g1[4] = (int)(tile1);                             // tile_dim1 | tile_dim2=0
    g1[5] = (int)(ld);                                // tensor_dim0_stride[31:0]
    g1[6] = 0;
    g1[7] = 0;
    v4i z4 = {};
#if defined(__has_include) && __has_include(<hip/amd_detail/amd_gfx1250_TDM.h>)
    v8i z8 = {};
    __builtin_amdgcn_tensor_load_to_lds(g0, g1, z4, z4, z8, 0);
#else
    __builtin_amdgcn_tensor_load_to_lds(g0, g1, z4, z4, 0);
#endif
}

// ---------- kernel 0: vectorized f32 -> bf16 conversion (8 elems/thread) ----------
__global__ void convert_f32_bf16(const float* __restrict__ src,
                                 unsigned short* __restrict__ dst, int n8) {
    int i = blockIdx.x * blockDim.x + threadIdx.x;
    if (i >= n8) return;
    const float4* sp = (const float4*)(src + (size_t)i * 8);
    float4 a = sp[0], b = sp[1];
    uint4 o = make_uint4(pack2bf(a.x, a.y), pack2bf(a.z, a.w),
                         pack2bf(b.x, b.y), pack2bf(b.z, b.w));
    *(uint4*)(dst + (size_t)i * 8) = o;
}

// ---------- kernel 1 (fast): h = leaky(xb @ W1b^T + b1) with pure TDM staging ----
// grid: (TOKENS/128, D_HID/128), block: 256 threads (8 wave32)
__global__ void __launch_bounds__(256)
gemm1_tdm_kernel(const unsigned short* __restrict__ xb,
                 const unsigned short* __restrict__ W1b,
                 const float* __restrict__ b1,
                 unsigned short* __restrict__ hbuf) {
    __shared__ unsigned short ldsA[2][128 * 32];   // 2 x 8 KB  x tile (bf16)
    __shared__ unsigned short ldsB[2][128 * 32];   // 2 x 8 KB  W1 tile (bf16)

    const int t     = threadIdx.x;
    const int lane  = t & 31;
    const int wid   = t >> 5;
    const int l16   = lane & 15;
    const int lhalf = lane >> 4;
    const int wm    = wid & 3;
    const int wn    = wid >> 2;

    const int tileM = blockIdx.x * 128;
    const int tileN = blockIdx.y * 128;

    const unsigned ldsAoff[2] = { (unsigned)(uintptr_t)&ldsA[0][0],
                                  (unsigned)(uintptr_t)&ldsA[1][0] };
    const unsigned ldsBoff[2] = { (unsigned)(uintptr_t)&ldsB[0][0],
                                  (unsigned)(uintptr_t)&ldsB[1][0] };

    const unsigned short* xbase  = xb  + (size_t)tileM * D_IN;
    const unsigned short* w1base = W1b + (size_t)tileN * D_IN;

    v8f c[2][4] = {};

    // prologue: DMA k0=0 into buffer 0 (wave0 -> A, wave1 -> B)
    if (wid == 0) tdm_load_2d_bf16(ldsAoff[0], xbase,  D_IN, TOKENS, 32, 128, D_IN);
    if (wid == 1) tdm_load_2d_bf16(ldsBoff[0], w1base, D_IN, D_HID,  32, 128, D_IN);
    __builtin_amdgcn_s_wait_tensorcnt(0);
    __syncthreads();

    for (int k0 = 0; k0 < D_IN; k0 += 32) {
        const int cur  = (k0 >> 5) & 1;
        const int nxt  = cur ^ 1;
        const bool more = (k0 + 32) < D_IN;

        if (more) {
            if (wid == 0) tdm_load_2d_bf16(ldsAoff[nxt], xbase  + (k0 + 32),
                                           D_IN, TOKENS, 32, 128, D_IN);
            if (wid == 1) tdm_load_2d_bf16(ldsBoff[nxt], w1base + (k0 + 32),
                                           D_IN, D_HID,  32, 128, D_IN);
        }

        Frag a[2], b[4];
#pragma unroll
        for (int mi = 0; mi < 2; ++mi) {
            const int rA = wm * 32 + mi * 16 + l16;
            const int koff = lhalf ? 8 : 0;
            a[mi].u[0] = *(const uint4*)&ldsA[cur][rA * 32 + koff];
            a[mi].u[1] = *(const uint4*)&ldsA[cur][rA * 32 + koff + 16];
        }
#pragma unroll
        for (int ni = 0; ni < 4; ++ni) {
            const int rB = wn * 64 + ni * 16 + l16;
            const int koff = lhalf * 8;
            const uint4* bp = (const uint4*)&ldsB[cur][rB * 32 + koff];
            b[ni].u[0] = bp[0];
            b[ni].u[1] = bp[2];
        }
#pragma unroll
        for (int mi = 0; mi < 2; ++mi)
#pragma unroll
            for (int ni = 0; ni < 4; ++ni)
                c[mi][ni] = __builtin_amdgcn_wmma_f32_16x16x32_bf16(
                    false, a[mi].v, false, b[ni].v,
                    (short)0, c[mi][ni], false, false);

        __builtin_amdgcn_s_wait_tensorcnt(0);
        __syncthreads();
    }

#pragma unroll
    for (int ni = 0; ni < 4; ++ni) {
        const int n = tileN + wn * 64 + ni * 16 + l16;
        const float bias = b1[n];
#pragma unroll
        for (int mi = 0; mi < 2; ++mi) {
            const int mbase = tileM + wm * 32 + mi * 16 + lhalf * 8;
#pragma unroll
            for (int v = 0; v < 8; ++v) {
                float val = c[mi][ni][v] + bias;
                val = (val >= 0.0f) ? val : SLOPE * val;
                hbuf[(size_t)(mbase + v) * D_HID + n] = f2bf(val);
            }
        }
    }
}

// ---------- kernel 1 (fallback, small ws): in-loop x conversion, TDM for W1 ----
__global__ void __launch_bounds__(256)
gemm1_cvt_kernel(const float* __restrict__ x,
                 const unsigned short* __restrict__ W1b,
                 const float* __restrict__ b1,
                 unsigned short* __restrict__ hbuf) {
    __shared__ unsigned short ldsA[2][128 * 32];
    __shared__ unsigned short ldsB[2][128 * 32];

    const int t     = threadIdx.x;
    const int lane  = t & 31;
    const int wid   = t >> 5;
    const int l16   = lane & 15;
    const int lhalf = lane >> 4;
    const int wm    = wid & 3;
    const int wn    = wid >> 2;

    const int tileM = blockIdx.x * 128;
    const int tileN = blockIdx.y * 128;

    const int srow  = t >> 1;
    const int shalf = t & 1;

    const unsigned ldsBoff[2] = { (unsigned)(uintptr_t)&ldsB[0][0],
                                  (unsigned)(uintptr_t)&ldsB[1][0] };

    v8f c[2][4] = {};
    float4 pf[4];

    if (wid == 0)
        tdm_load_2d_bf16(ldsBoff[0], W1b + (size_t)tileN * D_IN, D_IN, D_HID, 32, 128, D_IN);
    {
        const float* xp = x + (size_t)(tileM + srow) * D_IN + shalf * 16;
#pragma unroll
        for (int i = 0; i < 4; ++i) pf[i] = ((const float4*)xp)[i];
        uint4* dst = (uint4*)&ldsA[0][srow * 32 + shalf * 16];
        dst[0] = make_uint4(pack2bf(pf[0].x, pf[0].y), pack2bf(pf[0].z, pf[0].w),
                            pack2bf(pf[1].x, pf[1].y), pack2bf(pf[1].z, pf[1].w));
        dst[1] = make_uint4(pack2bf(pf[2].x, pf[2].y), pack2bf(pf[2].z, pf[2].w),
                            pack2bf(pf[3].x, pf[3].y), pack2bf(pf[3].z, pf[3].w));
    }
    __builtin_amdgcn_s_wait_tensorcnt(0);
    __syncthreads();

    for (int k0 = 0; k0 < D_IN; k0 += 32) {
        const int cur  = (k0 >> 5) & 1;
        const int nxt  = cur ^ 1;
        const bool more = (k0 + 32) < D_IN;

        if (more) {
            if (wid == 0)
                tdm_load_2d_bf16(ldsBoff[nxt], W1b + (size_t)tileN * D_IN + (k0 + 32),
                                 D_IN, D_HID, 32, 128, D_IN);
            const float* xp = x + (size_t)(tileM + srow) * D_IN + (k0 + 32) + shalf * 16;
#pragma unroll
            for (int i = 0; i < 4; ++i) pf[i] = ((const float4*)xp)[i];
        }

        Frag a[2], b[4];
#pragma unroll
        for (int mi = 0; mi < 2; ++mi) {
            const int rA = wm * 32 + mi * 16 + l16;
            const int koff = lhalf ? 8 : 0;
            a[mi].u[0] = *(const uint4*)&ldsA[cur][rA * 32 + koff];
            a[mi].u[1] = *(const uint4*)&ldsA[cur][rA * 32 + koff + 16];
        }
#pragma unroll
        for (int ni = 0; ni < 4; ++ni) {
            const int rB = wn * 64 + ni * 16 + l16;
            const int koff = lhalf * 8;
            const uint4* bp = (const uint4*)&ldsB[cur][rB * 32 + koff];
            b[ni].u[0] = bp[0];
            b[ni].u[1] = bp[2];
        }
#pragma unroll
        for (int mi = 0; mi < 2; ++mi)
#pragma unroll
            for (int ni = 0; ni < 4; ++ni)
                c[mi][ni] = __builtin_amdgcn_wmma_f32_16x16x32_bf16(
                    false, a[mi].v, false, b[ni].v,
                    (short)0, c[mi][ni], false, false);

        if (more) {
            uint4* dst = (uint4*)&ldsA[nxt][srow * 32 + shalf * 16];
            dst[0] = make_uint4(pack2bf(pf[0].x, pf[0].y), pack2bf(pf[0].z, pf[0].w),
                                pack2bf(pf[1].x, pf[1].y), pack2bf(pf[1].z, pf[1].w));
            dst[1] = make_uint4(pack2bf(pf[2].x, pf[2].y), pack2bf(pf[2].z, pf[2].w),
                                pack2bf(pf[3].x, pf[3].y), pack2bf(pf[3].z, pf[3].w));
        }
        __builtin_amdgcn_s_wait_tensorcnt(0);
        __syncthreads();
    }

#pragma unroll
    for (int ni = 0; ni < 4; ++ni) {
        const int n = tileN + wn * 64 + ni * 16 + l16;
        const float bias = b1[n];
#pragma unroll
        for (int mi = 0; mi < 2; ++mi) {
            const int mbase = tileM + wm * 32 + mi * 16 + lhalf * 8;
#pragma unroll
            for (int v = 0; v < 8; ++v) {
                float val = c[mi][ni][v] + bias;
                val = (val >= 0.0f) ? val : SLOPE * val;
                hbuf[(size_t)(mbase + v) * D_HID + n] = f2bf(val);
            }
        }
    }
}

// ---------- kernel 2: logits = h @ W2^T + b2; top-8 + softmax (TDM staged) -----
// grid: TOKENS/128, block: 256 threads (8 wave32, 16 tokens per wave)
__global__ void __launch_bounds__(256)
gemm2_topk_kernel(const unsigned short* __restrict__ hbuf,
                  const unsigned short* __restrict__ W2b,
                  const float* __restrict__ b2,
                  float* __restrict__ probs,
                  int* __restrict__ inds) {
    __shared__ unsigned short ldsA[2][128 * 32];   // 2 x 8 KB  h tile
    __shared__ unsigned short ldsB[2][64 * 32];    // 2 x 4 KB  W2 tile
    __shared__ float lsc[8][16][N_EXP + 1];        // 33 KB logits scratch

    const int t     = threadIdx.x;
    const int lane  = t & 31;
    const int wid   = t >> 5;
    const int l16   = lane & 15;
    const int lhalf = lane >> 4;

    const int blkTok = blockIdx.x * 128;
    const int tBase  = blkTok + wid * 16;

    const unsigned ldsAoff[2] = { (unsigned)(uintptr_t)&ldsA[0][0],
                                  (unsigned)(uintptr_t)&ldsA[1][0] };
    const unsigned ldsBoff[2] = { (unsigned)(uintptr_t)&ldsB[0][0],
                                  (unsigned)(uintptr_t)&ldsB[1][0] };

    const unsigned short* hbase = hbuf + (size_t)blkTok * D_HID;

    v8f c[4] = {};

    if (wid == 0) tdm_load_2d_bf16(ldsAoff[0], hbase, D_HID, TOKENS, 32, 128, D_HID);
    if (wid == 1) tdm_load_2d_bf16(ldsBoff[0], W2b,   D_HID, N_EXP,  32, 64,  D_HID);
    __builtin_amdgcn_s_wait_tensorcnt(0);
    __syncthreads();

    for (int k0 = 0; k0 < D_HID; k0 += 32) {
        const int cur  = (k0 >> 5) & 1;
        const int nxt  = cur ^ 1;
        const bool more = (k0 + 32) < D_HID;

        if (more) {
            if (wid == 0) tdm_load_2d_bf16(ldsAoff[nxt], hbase + (k0 + 32),
                                           D_HID, TOKENS, 32, 128, D_HID);
            if (wid == 1) tdm_load_2d_bf16(ldsBoff[nxt], W2b + (k0 + 32),
                                           D_HID, N_EXP,  32, 64,  D_HID);
        }

        Frag a;
        {
            const int rA = wid * 16 + l16;
            const int koff = lhalf ? 8 : 0;
            a.u[0] = *(const uint4*)&ldsA[cur][rA * 32 + koff];
            a.u[1] = *(const uint4*)&ldsA[cur][rA * 32 + koff + 16];
        }
#pragma unroll
        for (int ni = 0; ni < 4; ++ni) {
            Frag b;
            const int rB = ni * 16 + l16;
            const int koff = lhalf * 8;
            const uint4* bp = (const uint4*)&ldsB[cur][rB * 32 + koff];
            b.u[0] = bp[0];
            b.u[1] = bp[2];
            c[ni] = __builtin_amdgcn_wmma_f32_16x16x32_bf16(
                false, a.v, false, b.v, (short)0, c[ni], false, false);
        }
        __builtin_amdgcn_s_wait_tensorcnt(0);
        __syncthreads();
    }

#pragma unroll
    for (int ni = 0; ni < 4; ++ni) {
        const int e = ni * 16 + l16;
        const float bias = b2[e];
#pragma unroll
        for (int v = 0; v < 8; ++v) {
            const int tokLocal = v + lhalf * 8;
            lsc[wid][tokLocal][e] = c[ni][v] + bias;
        }
    }
    __syncthreads();

    if (lane < 16) {
        float* row = lsc[wid][lane];
        const int tok = tBase + lane;
        float vals[TOP_K];
        int   idxs[TOP_K];
#pragma unroll
        for (int i = 0; i < TOP_K; ++i) {
            float best = -3.402823466e38f;
            int bi = 0;
            for (int e = 0; e < N_EXP; ++e) {
                float xv = row[e];
                if (xv > best) { best = xv; bi = e; }
            }
            vals[i] = best;
            idxs[i] = bi;
            row[bi] = -3.402823466e38f;
        }
        float sum = 0.0f;
        float ex[TOP_K];
#pragma unroll
        for (int i = 0; i < TOP_K; ++i) {
            ex[i] = __expf(vals[i] - vals[0]);
            sum += ex[i];
        }
        const float inv = 1.0f / sum;
#pragma unroll
        for (int i = 0; i < TOP_K; ++i) {
            probs[(size_t)tok * TOP_K + i] = ex[i] * inv;
            inds [(size_t)tok * TOP_K + i] = idxs[i];
        }
    }
}

// ---------- host launcher ----------
extern "C" void kernel_launch(void* const* d_in, const int* in_sizes, int n_in,
                              void* d_out, int out_size, void* d_ws, size_t ws_size,
                              hipStream_t stream) {
    const float* x  = (const float*)d_in[0];
    const float* W1 = (const float*)d_in[1];
    const float* b1 = (const float*)d_in[2];
    const float* W2 = (const float*)d_in[3];
    const float* b2 = (const float*)d_in[4];

    unsigned char* ws = (unsigned char*)d_ws;
    const size_t W1B_BYTES = (size_t)D_HID * D_IN * 2;      //  4,194,304
    const size_t W2B_BYTES = (size_t)N_EXP * D_HID * 2;     //    131,072
    const size_t XB_BYTES  = (size_t)TOKENS * D_IN * 2;     // 67,108,864
    const size_t HB_BYTES  = (size_t)TOKENS * D_HID * 2;    // 33,554,432

    unsigned short* W1b = (unsigned short*)(ws);
    unsigned short* W2b = (unsigned short*)(ws + W1B_BYTES);

    float* probs = (float*)d_out;
    int*   inds  = (int*)d_out + TOKENS * TOP_K;

    const bool fast = ws_size >= (W1B_BYTES + W2B_BYTES + XB_BYTES + HB_BYTES);

    // weight conversions (always)
    convert_f32_bf16<<<(D_HID * D_IN / 8 + 255) / 256, 256, 0, stream>>>(W1, W1b, D_HID * D_IN / 8);
    convert_f32_bf16<<<(N_EXP * D_HID / 8 + 255) / 256, 256, 0, stream>>>(W2, W2b, N_EXP * D_HID / 8);

    if (fast) {
        unsigned short* xb = (unsigned short*)(ws + W1B_BYTES + W2B_BYTES);
        unsigned short* hb = (unsigned short*)(ws + W1B_BYTES + W2B_BYTES + XB_BYTES);
        convert_f32_bf16<<<(TOKENS * D_IN / 8 + 255) / 256, 256, 0, stream>>>(x, xb, TOKENS * D_IN / 8);
        dim3 grid(TOKENS / 128, D_HID / 128);
        gemm1_tdm_kernel<<<grid, 256, 0, stream>>>(xb, W1b, b1, hb);
        gemm2_topk_kernel<<<TOKENS / 128, 256, 0, stream>>>(hb, W2b, b2, probs, inds);
    } else {
        unsigned short* hb = (unsigned short*)(ws + W1B_BYTES + W2B_BYTES);
        dim3 grid(TOKENS / 128, D_HID / 128);
        gemm1_cvt_kernel<<<grid, 256, 0, stream>>>(x, W1b, b1, hb);
        gemm2_topk_kernel<<<TOKENS / 128, 256, 0, stream>>>(hb, W2b, b2, probs, inds);
    }
}
